// CodebookQuantizer_80247168959076
// MI455X (gfx1250) — matrix operand order repaired and checked by reading, not compiled, not run on hardware
//
#include <hip/hip_runtime.h>
#include <stdint.h>

typedef __bf16 bf16_t;
typedef __attribute__((ext_vector_type(16))) __bf16 v16bf;
typedef __attribute__((ext_vector_type(8)))  float  v8f;

#define B_ 4
#define Q_ 4096
#define D_ 512
#define K_ 8192
#define TEMP_ 0.1f
#define LOG2E_ 1.4426950408889634f
#define S2_ (2.0f * LOG2E_ / TEMP_)

union Frag { v16bf v; int4 q[2]; };

static __device__ __forceinline__ uint16_t f2bf(float f) {
  return __builtin_bit_cast(uint16_t, (bf16_t)f);
}

// ---- Prologue A: codebook f32 -> bf16 (row major) + scaled ||e||^2 ----
__global__ void cvtA_kernel(const float* __restrict__ cb,
                            uint16_t* __restrict__ g_cb,
                            float* __restrict__ esq) {
  const int k = blockIdx.x;   // 0..K-1
  const int t = threadIdx.x;  // 0..255
  float s = 0.0f;
#pragma unroll
  for (int i = 0; i < 2; ++i) {
    const int d = t + 256 * i;
    const float v = cb[(size_t)k * D_ + d];
    s += v * v;
    g_cb[(size_t)k * D_ + d] = f2bf(v);
  }
#pragma unroll
  for (int m = 1; m < 32; m <<= 1) s += __shfl_xor(s, m, 32);
  __shared__ float red[8];
  if ((t & 31) == 0) red[t >> 5] = s;
  __syncthreads();
  if (t == 0) {
    float tot = 0.0f;
#pragma unroll
    for (int w = 0; w < 8; ++w) tot += red[w];
    esq[k] = tot * (LOG2E_ / TEMP_);
  }
}

// ---- Prologue B: codebook f32 -> bf16 transposed [D][K] via LDS tiles ----
__global__ void cvtB_kernel(const float* __restrict__ cb,
                            uint16_t* __restrict__ g_cbT) {
  __shared__ uint16_t tile[64][65];
  const int t = threadIdx.x;
  const int k0 = (blockIdx.x >> 3) << 6;  // 128 k-tiles
  const int d0 = (blockIdx.x & 7) << 6;   // 8 d-tiles
#pragma unroll
  for (int s = 0; s < 16; ++s) {
    const int i = t + 256 * s;
    const int kk = i >> 6, dd = i & 63;
    tile[kk][dd] = f2bf(cb[(size_t)(k0 + kk) * D_ + d0 + dd]);
  }
  __syncthreads();
#pragma unroll
  for (int s = 0; s < 16; ++s) {
    const int j = t + 256 * s;
    const int dd = j >> 6, kk = j & 63;
    g_cbT[(size_t)(d0 + dd) * K_ + k0 + kk] = tile[kk][dd];
  }
}

// Issue async (ASYNCcnt-tracked) global->LDS staging for one 32-k codebook
// chunk: 8x b128 per thread into cb_buf ([k][d]) + 8x into cbT_buf ([d][k]).
static __device__ __forceinline__ void stage_async(int tid, int nkc,
                                                   uint16_t* cb_buf,
                                                   uint16_t* cbT_buf,
                                                   const uint16_t* g_cb,
                                                   const uint16_t* g_cbT) {
  const uint32_t lds_cb_base  = (uint32_t)(uintptr_t)cb_buf;
  const uint32_t lds_cbT_base = (uint32_t)(uintptr_t)cbT_buf;
  const uint64_t gcb  = (uint64_t)(uintptr_t)g_cb;
  const uint64_t gcbT = (uint64_t)(uintptr_t)g_cbT;
#pragma unroll
  for (int s = 0; s < 8; ++s) {
    const uint32_t i = (uint32_t)(tid + 256 * s);        // 0..2047 b128 units
    const uint32_t lds_a = lds_cb_base + i * 16u;
    const uint32_t goff  = (uint32_t)nkc * 1024u + i * 16u;  // chunk is contiguous
    asm volatile("global_load_async_to_lds_b128 %0, %1, %2"
                 :: "v"(lds_a), "v"(goff), "s"(gcb) : "memory");
  }
#pragma unroll
  for (int s = 0; s < 8; ++s) {
    const uint32_t i = (uint32_t)(tid + 256 * s);        // 0..2047
    const uint32_t d = i >> 2, piece = i & 3;            // 4x b128 per d-row of 32 k
    const uint32_t lds_a = lds_cbT_base + i * 16u;
    const uint32_t goff  = (d * (uint32_t)K_ + (uint32_t)nkc) * 2u + piece * 16u;
    asm volatile("global_load_async_to_lds_b128 %0, %1, %2"
                 :: "v"(lds_a), "v"(goff), "s"(gcbT) : "memory");
  }
}

// ---- Main fused kernel: scores -> online softmax -> weighted codebook sum ----
__launch_bounds__(256)
__global__ void vq_kernel(const float* __restrict__ hmat,
                          const uint16_t* __restrict__ g_cb,
                          const uint16_t* __restrict__ g_cbT,
                          const float* __restrict__ esq,
                          float* __restrict__ out) {
  __shared__ __align__(16) uint16_t lds_cb[2][32 * 512];   // [k][d] bf16, 2x32KB
  __shared__ __align__(16) uint16_t lds_cbT[2][512 * 32];  // [d][k] bf16, 2x32KB
  __shared__ __align__(16) uint16_t p_lds[64 * 32];        // exp(P) bf16, 4KB
  __shared__ float tmax[64][2];
  __shared__ float tsum[64][2];

  const int tid  = threadIdx.x;
  const int wave = tid >> 5;
  const int lane = tid & 31;
  const int r    = wave >> 1;   // q-row block (0..3)
  const int c    = wave & 1;    // k-sub-chunk for scores / D-half for output
  const int lo   = lane & 15;
  const int hi   = lane >> 4;

  const size_t rowBase = (size_t)blockIdx.x * 64;  // flattened (b*Q + q)

  // ---- h tile (16 rows x 512) as 16 bf16 A-fragments, kept in registers ----
  v16bf ha[16];
  {
    const float* hrow = hmat + (rowBase + (size_t)(r * 16 + lo)) * D_;
    const int off = hi * 16;
#pragma unroll
    for (int dc = 0; dc < 16; ++dc) {
      const float4 q0 = *(const float4*)(hrow + dc * 32 + off);
      const float4 q1 = *(const float4*)(hrow + dc * 32 + off + 4);
      const float4 q2 = *(const float4*)(hrow + dc * 32 + off + 8);
      const float4 q3 = *(const float4*)(hrow + dc * 32 + off + 12);
      v16bf a;
      a[0]=(bf16_t)q0.x;  a[1]=(bf16_t)q0.y;  a[2]=(bf16_t)q0.z;  a[3]=(bf16_t)q0.w;
      a[4]=(bf16_t)q1.x;  a[5]=(bf16_t)q1.y;  a[6]=(bf16_t)q1.z;  a[7]=(bf16_t)q1.w;
      a[8]=(bf16_t)q2.x;  a[9]=(bf16_t)q2.y;  a[10]=(bf16_t)q2.z; a[11]=(bf16_t)q2.w;
      a[12]=(bf16_t)q3.x; a[13]=(bf16_t)q3.y; a[14]=(bf16_t)q3.z; a[15]=(bf16_t)q3.w;
      ha[dc] = a;
    }
  }

  // O accumulators: 16 tiles of 16x16 f32 covering this wave's 256 d-columns
  v8f o[16];
#pragma unroll
  for (int t = 0; t < 16; ++t) {
#pragma unroll
    for (int e = 0; e < 8; ++e) o[t][e] = 0.0f;
  }

  float mstate[8], lstate[8];
#pragma unroll
  for (int j = 0; j < 8; ++j) { mstate[j] = -3.0e38f; lstate[j] = 0.0f; }

  // Pre-stage chunk 0 into buffer 0 (async).
  stage_async(tid, 0, lds_cb[0], lds_cbT[0], g_cb, g_cbT);

  for (int kc = 0; kc < K_; kc += 32) {
    const int buf = (kc >> 5) & 1;
    __syncthreads();  // all readers of lds_*[buf^1] (iter kc-32) are done

    // Prefetch next chunk into the other buffer (wrap at the end: harmless).
    const int nkc = (kc + 32) & (K_ - 1);
    stage_async(tid, nkc, lds_cb[buf ^ 1], lds_cbT[buf ^ 1], g_cb, g_cbT);

    // Retire this chunk's async batch (first 16 of <=32 outstanding, in-order),
    // then make every wave's staged portion visible.
    asm volatile("s_wait_asynccnt 16" ::: "memory");
    __syncthreads();

    const uint16_t* cbB  = lds_cb[buf];
    const uint16_t* cbTB = lds_cbT[buf];

    // ---- scores: S[16q x 16k] = h . cb^T over D=512 (16 WMMAs) ----
    // Ping-pong B fragments so >=2 ds_loads stay in flight per WMMA.
    v8f sf;
#pragma unroll
    for (int e = 0; e < 8; ++e) sf[e] = 0.0f;
    {
      const uint16_t* bbase = cbB + (c * 16 + lo) * 512 + hi * 16;
      Frag f0, f1;
      f0.q[0] = *(const int4*)(bbase);
      f0.q[1] = *(const int4*)(bbase + 8);
#pragma unroll
      for (int dc = 0; dc < 16; dc += 2) {
        const uint16_t* b1 = bbase + (dc + 1) * 32;
        f1.q[0] = *(const int4*)(b1);
        f1.q[1] = *(const int4*)(b1 + 8);
        sf = __builtin_amdgcn_wmma_f32_16x16x32_bf16(false, ha[dc], false, f0.v,
                                                     (short)0, sf, false, false);
        if (dc + 2 < 16) {
          const uint16_t* b2 = bbase + (dc + 2) * 32;
          f0.q[0] = *(const int4*)(b2);
          f0.q[1] = *(const int4*)(b2 + 8);
        }
        sf = __builtin_amdgcn_wmma_f32_16x16x32_bf16(false, ha[dc + 1], false, f1.v,
                                                     (short)0, sf, false, false);
      }
    }

    // logits in base-2: x = 2*cross*log2e/T - e_sq*log2e/T  (||h||^2 cancels)
    const float eq = esq[kc + c * 16 + lo];
    float x[8], mt[8];
#pragma unroll
    for (int j = 0; j < 8; ++j) { x[j] = sf[j] * S2_ - eq; mt[j] = x[j]; }
    // tile row-max across this wave's 16 columns (reduce within 16-lane half)
#pragma unroll
    for (int m = 1; m < 16; m <<= 1) {
#pragma unroll
      for (int j = 0; j < 8; ++j) mt[j] = fmaxf(mt[j], __shfl_xor(mt[j], m, 32));
    }
    if (lo == 0) {
#pragma unroll
      for (int j = 0; j < 8; ++j) tmax[r * 16 + j + 8 * hi][c] = mt[j];
    }
    __syncthreads();

    // combine both k-sub-tiles -> consistent m_new / alpha across D-half waves
    float alpha[8];
    bool nochange = true;
#pragma unroll
    for (int j = 0; j < 8; ++j) {
      const int row = r * 16 + j + 8 * hi;
      const float cm = fmaxf(tmax[row][0], tmax[row][1]);
      const float mnew = fmaxf(mstate[j], cm);
      alpha[j] = exp2f(mstate[j] - mnew);
      nochange = nochange && (mnew == mstate[j]);
      x[j] = exp2f(x[j] - mnew);  // x now holds p
      mstate[j] = mnew;
    }
#pragma unroll
    for (int j = 0; j < 8; ++j)
      p_lds[(r * 16 + j + 8 * hi) * 32 + c * 16 + lo] = f2bf(x[j]);
    // tile row-sum of p
#pragma unroll
    for (int m = 1; m < 16; m <<= 1) {
#pragma unroll
      for (int j = 0; j < 8; ++j) x[j] += __shfl_xor(x[j], m, 32);
    }
    if (lo == 0) {
#pragma unroll
      for (int j = 0; j < 8; ++j) tsum[r * 16 + j + 8 * hi][c] = x[j];
    }
    __syncthreads();

#pragma unroll
    for (int j = 0; j < 8; ++j) {
      const int row = r * 16 + j + 8 * hi;
      lstate[j] = lstate[j] * alpha[j] + tsum[row][0] + tsum[row][1];
    }
    // rescale running O by alpha (per-row); skip when every row's max is
    // unchanged in this whole wave (alpha==1 exactly) -- common once the
    // peaked (T=0.1) softmax max stabilizes.
    if (!__all(nochange)) {
#pragma unroll
      for (int t = 0; t < 16; ++t) {
#pragma unroll
        for (int j = 0; j < 8; ++j) o[t][j] *= alpha[j];
      }
    }

    // ---- O[16q x 256d] += P[16q x 32k] @ CB[32k x 256d] (16 WMMAs) ----
    Frag pa;
    {
      const uint16_t* pp = p_lds + lo * 32 + hi * 16;
      pa.q[0] = *(const int4*)(pp);
      pa.q[1] = *(const int4*)(pp + 8);
    }
    {
      const uint16_t* obase = cbTB + (c * 256 + lo) * 32 + hi * 16;
      Frag g0, g1;
      g0.q[0] = *(const int4*)(obase);
      g0.q[1] = *(const int4*)(obase + 8);
#pragma unroll
      for (int t = 0; t < 16; t += 2) {
        const uint16_t* b1 = obase + (t + 1) * (16 * 32);
        g1.q[0] = *(const int4*)(b1);
        g1.q[1] = *(const int4*)(b1 + 8);
        o[t] = __builtin_amdgcn_wmma_f32_16x16x32_bf16(false, pa.v, false, g0.v,
                                                       (short)0, o[t], false, false);
        if (t + 2 < 16) {
          const uint16_t* b2 = obase + (t + 2) * (16 * 32);
          g0.q[0] = *(const int4*)(b2);
          g0.q[1] = *(const int4*)(b2 + 8);
        }
        o[t + 1] = __builtin_amdgcn_wmma_f32_16x16x32_bf16(false, pa.v, false, g1.v,
                                                           (short)0, o[t + 1], false, false);
      }
    }
  }

  // ---- normalize by softmax denominator and store z_e (forward of z_q) ----
  float inv[8];
#pragma unroll
  for (int j = 0; j < 8; ++j) inv[j] = 1.0f / lstate[j];
#pragma unroll
  for (int t = 0; t < 16; ++t) {
#pragma unroll
    for (int j = 0; j < 8; ++j) {
      const size_t row = rowBase + (size_t)(r * 16 + j + 8 * hi);
      out[row * D_ + c * 256 + t * 16 + lo] = o[t][j] * inv[j];
    }
  }
}

extern "C" void kernel_launch(void* const* d_in, const int* in_sizes, int n_in,
                              void* d_out, int out_size, void* d_ws, size_t ws_size,
                              hipStream_t stream) {
  (void)in_sizes; (void)n_in; (void)out_size; (void)ws_size;
  const float* h  = (const float*)d_in[0];
  const float* cb = (const float*)d_in[1];
  // workspace carve-out: cb_bf16 (8MB) | cbT_bf16 (8MB) | esq_scaled (32KB)
  uint16_t* g_cb  = (uint16_t*)d_ws;
  uint16_t* g_cbT = g_cb + (size_t)K_ * D_;
  float*    esq   = (float*)(g_cbT + (size_t)D_ * K_);

  cvtA_kernel<<<K_, 256, 0, stream>>>(cb, g_cb, esq);
  cvtB_kernel<<<(K_ / 64) * (D_ / 64), 256, 0, stream>>>(cb, g_cbT);
  vq_kernel<<<(B_ * Q_) / 64, 256, 0, stream>>>(h, g_cb, g_cbT, esq, (float*)d_out);
}
